// TogepiMultiHeadAttention_23218593202680
// MI455X (gfx1250) — matrix operand order repaired and checked by reading, not compiled, not run on hardware
//
#include <hip/hip_runtime.h>
#include <hip/hip_bf16.h>
#include <math.h>

// ---------------------------------------------------------------------------
// Types
// ---------------------------------------------------------------------------
typedef __bf16 bf16_t;
typedef __attribute__((ext_vector_type(16))) __bf16 bf16x16;
typedef __attribute__((ext_vector_type(8)))  float  f32x8;
typedef unsigned int u32;
typedef unsigned long long u64;
typedef __attribute__((ext_vector_type(4))) u32 u32x4;
typedef __attribute__((ext_vector_type(4))) int i32x4;
typedef __attribute__((ext_vector_type(8))) int i32x8;

struct c2 { float x, y; };
__device__ inline c2 cadd(c2 a, c2 b) { return {a.x + b.x, a.y + b.y}; }
__device__ inline c2 csub(c2 a, c2 b) { return {a.x - b.x, a.y - b.y}; }
__device__ inline c2 cmul(c2 a, c2 b) { return {a.x * b.x - a.y * b.y, a.x * b.y + a.y * b.x}; }

#define BB 4
#define LL 2048
#define EE 1024
#define HH 16
#define DD 64
#define NSEQ 8192      // padded FFT length (linear conv 8189 <= 8192)
#define NTAP 4095      // 2L-1

// ---------------------------------------------------------------------------
// f32 -> bf16 conversion (grid-stride)
// ---------------------------------------------------------------------------
__global__ __launch_bounds__(256) void cvt_f32_bf16_kernel(const float* __restrict__ src,
                                                           bf16_t* __restrict__ dst,
                                                           long long n) {
  long long i = (long long)blockIdx.x * blockDim.x + threadIdx.x;
  long long stride = (long long)gridDim.x * blockDim.x;
  for (; i < n; i += stride) dst[i] = (bf16_t)src[i];
}

// sparse (L x L) -> bf16 with lower-triangular mask applied
__global__ __launch_bounds__(256) void cvt_tril_bf16_kernel(const float* __restrict__ src,
                                                            bf16_t* __restrict__ dst) {
  long long n = (long long)LL * LL;
  long long i = (long long)blockIdx.x * blockDim.x + threadIdx.x;
  long long stride = (long long)gridDim.x * blockDim.x;
  for (; i < n; i += stride) {
    int l = (int)(i >> 11);
    int s = (int)(i & (LL - 1));
    dst[i] = (s <= l) ? (bf16_t)src[i] : (bf16_t)0.0f;
  }
}

// ---------------------------------------------------------------------------
// PSF softmax over 4095 taps (taps >= L have masked logit == 0)
// ---------------------------------------------------------------------------
__global__ __launch_bounds__(256) void psf_softmax_kernel(const float* __restrict__ w,
                                                          float* __restrict__ out) {
  __shared__ float red[256];
  int h = blockIdx.x >> 6, d = blockIdx.x & 63;
  const float* col = w + (size_t)h * ((size_t)NTAP * DD) + d;
  float* ocol = out + (size_t)h * ((size_t)NTAP * DD) + d;

  float mx = 0.0f;  // masked logits are exactly 0, include in max
  for (int t = threadIdx.x; t < LL; t += 256) mx = fmaxf(mx, col[(size_t)t * DD]);
  red[threadIdx.x] = mx; __syncthreads();
  for (int st = 128; st > 0; st >>= 1) {
    if (threadIdx.x < st) red[threadIdx.x] = fmaxf(red[threadIdx.x], red[threadIdx.x + st]);
    __syncthreads();
  }
  mx = red[0]; __syncthreads();

  float sum = 0.0f;
  for (int t = threadIdx.x; t < NTAP; t += 256) {
    float lg = (t < LL) ? col[(size_t)t * DD] : 0.0f;
    sum += __expf(lg - mx);
  }
  red[threadIdx.x] = sum; __syncthreads();
  for (int st = 128; st > 0; st >>= 1) {
    if (threadIdx.x < st) red[threadIdx.x] += red[threadIdx.x + st];
    __syncthreads();
  }
  float inv = 1.0f / red[0];
  for (int t = threadIdx.x; t < NTAP; t += 256) {
    float lg = (t < LL) ? col[(size_t)t * DD] : 0.0f;
    ocol[(size_t)t * DD] = __expf(lg - mx) * inv;
  }
}

// ---------------------------------------------------------------------------
// serial per-thread 64-point complex FFT on a private LDS chunk (stride 1)
// ---------------------------------------------------------------------------
__device__ void fft64_serial(c2* a, float sign) {
  for (int i = 0; i < 64; ++i) {
    int j = (int)(__brev((unsigned)i) >> 26);
    if (j > i) { c2 t = a[i]; a[i] = a[j]; a[j] = t; }
  }
  for (int len = 2; len <= 64; len <<= 1) {
    int half = len >> 1;
    float ang0 = sign * 6.28318530718f / (float)len;
    for (int i = 0; i < 64; i += len) {
      for (int j = 0; j < half; ++j) {
        float sn, cs; __sincosf(ang0 * (float)j, &sn, &cs);
        c2 w = {cs, sn};
        c2 u = a[i + j];
        c2 v = cmul(a[i + j + half], w);
        a[i + j] = cadd(u, v);
        a[i + j + half] = csub(u, v);
      }
    }
  }
}

// D-axis forward FFT of real rows (serves both psf and emb via strides)
__global__ __launch_bounds__(64) void fft64_rows_kernel(const float* __restrict__ src,
                                                        long long headStride,
                                                        long long rowStride,
                                                        c2* __restrict__ dst,
                                                        int rowsPerHead) {
  __shared__ c2 sm[64 * 65];  // pad 65 to spread LDS banks
  int gid = blockIdx.x * 64 + threadIdx.x;
  int total = HH * rowsPerHead;
  if (gid >= total) return;  // no barriers in this kernel
  int h = gid / rowsPerHead;
  int t = gid - h * rowsPerHead;
  c2* a = sm + threadIdx.x * 65;
  const float* s = src + (size_t)h * headStride + (size_t)t * rowStride;
  for (int d = 0; d < 64; ++d) a[d] = {s[d], 0.0f};
  fft64_serial(a, -1.0f);
  c2* o = dst + ((size_t)h * rowsPerHead + t) * 64;
  for (int d = 0; d < 64; ++d) o[d] = a[d];
}

// ---------------------------------------------------------------------------
// cooperative in-place 8192-point complex FFT in LDS (256 threads)
// ---------------------------------------------------------------------------
__device__ void fft8192_inplace(c2* X, float sign) {
  for (int i = threadIdx.x; i < NSEQ; i += 256) {
    int j = (int)(__brev((unsigned)i) >> 19);
    if (j > i) { c2 t = X[i]; X[i] = X[j]; X[j] = t; }
  }
  __syncthreads();
  for (int st = 0; st < 13; ++st) {
    int half = 1 << st;
    int len = half << 1;
    float ang0 = sign * 6.28318530718f / (float)len;
    for (int idx = threadIdx.x; idx < (NSEQ >> 1); idx += 256) {
      int j = idx & (half - 1);
      int i0 = ((idx >> st) << (st + 1)) + j;
      float sn, cs; __sincosf(ang0 * (float)j, &sn, &cs);
      c2 w = {cs, sn};
      c2 u = X[i0];
      c2 v = cmul(X[i0 + half], w);
      X[i0] = cadd(u, v);
      X[i0 + half] = csub(u, v);
    }
    __syncthreads();
  }
}

// forward seq-FFT of PSF columns (already D-FFT'd); zero-pad 4095 -> 8192
__global__ __launch_bounds__(256) void psf_seqfft_kernel(const c2* __restrict__ psfRows,
                                                         c2* __restrict__ psfF) {
  __shared__ c2 X[NSEQ];  // 64 KB
  int h = blockIdx.x >> 6, d = blockIdx.x & 63;
  const c2* col = psfRows + (size_t)h * ((size_t)NTAP * DD) + d;
  for (int t = threadIdx.x; t < NSEQ; t += 256)
    X[t] = (t < NTAP) ? col[(size_t)t * DD] : c2{0.0f, 0.0f};
  __syncthreads();
  fft8192_inplace(X, -1.0f);
  c2* out = psfF + (size_t)h * ((size_t)NSEQ * DD) + d;
  for (int t = threadIdx.x; t < NSEQ; t += 256) out[(size_t)t * DD] = X[t];
}

// fused forward seq-FFT -> multiply by psf_fft -> inverse seq-FFT per column.
// keeps rows needed after fold: r<2048 -> t=r ; r>=2048 -> t=r+2047 (=n+4095)
__global__ __launch_bounds__(256) void emb_seqfft_kernel(const c2* __restrict__ embRows,
                                                         const c2* __restrict__ psfF,
                                                         c2* __restrict__ embG) {
  __shared__ c2 X[NSEQ];  // 64 KB
  int h = blockIdx.x >> 6, d = blockIdx.x & 63;
  const c2* col = embRows + (size_t)h * ((size_t)LL * DD) + d;
  for (int t = threadIdx.x; t < NSEQ; t += 256)
    X[t] = (t < LL) ? col[(size_t)t * DD] : c2{0.0f, 0.0f};
  __syncthreads();
  fft8192_inplace(X, -1.0f);
  const c2* pc = psfF + (size_t)h * ((size_t)NSEQ * DD) + d;
  for (int t = threadIdx.x; t < NSEQ; t += 256) X[t] = cmul(X[t], pc[(size_t)t * DD]);
  __syncthreads();
  fft8192_inplace(X, 1.0f);
  c2* g = embG + (size_t)h * ((size_t)2 * LL * DD) + d;
  for (int r = threadIdx.x; r < 2 * LL; r += 256) {
    int t = (r < LL) ? r : (r + NTAP - LL);  // r>=2048 -> t = r + 2047
    g[(size_t)r * DD] = X[t];
  }
}

// fold (circular-4095 from linear-8192), inverse D-axis FFT, real part, GELU
__global__ __launch_bounds__(64) void idfft_fold_gelu_kernel(const c2* __restrict__ embG,
                                                             float* __restrict__ conv) {
  __shared__ c2 sm[64 * 65];
  int h = blockIdx.y;
  int n = blockIdx.x * 64 + threadIdx.x;  // n in [0, 2048)
  c2* a = sm + threadIdx.x * 65;
  const c2* g = embG + (size_t)h * ((size_t)2 * LL * DD);
  const c2* r0 = g + (size_t)n * DD;
  const c2* r1 = g + (size_t)(n + LL) * DD;
  for (int d = 0; d < 64; ++d) a[d] = cadd(r0[d], r1[d]);  // fold commutes with d-IFFT
  fft64_serial(a, 1.0f);
  const float scale = 1.0f / ((float)NSEQ * (float)DD);
  float* o = conv + (size_t)n * EE + h * DD;
  for (int d = 0; d < 64; ++d) {
    float v = a[d].x * scale;
    o[d] = 0.5f * v * (1.0f + erff(v * 0.70710678118f));  // exact gelu
  }
}

// ---------------------------------------------------------------------------
// LDS-tiled transpose with bf16 convert: [L, E] f32 per batch -> [E, L] bf16
// ---------------------------------------------------------------------------
__global__ __launch_bounds__(256) void transpose_bf16_kernel(const float* __restrict__ src,
                                                             bf16_t* __restrict__ dst) {
  __shared__ float tile[32][33];
  int b = blockIdx.z;
  const float* s = src + (size_t)b * LL * EE;
  bf16_t* o = dst + (size_t)b * LL * EE;
  int l0 = blockIdx.x * 32, e0 = blockIdx.y * 32;
  int tx = threadIdx.x & 31, ty = threadIdx.x >> 5;  // 32 x 8
  for (int i = 0; i < 32; i += 8)
    tile[ty + i][tx] = s[(size_t)(l0 + ty + i) * EE + (e0 + tx)];
  __syncthreads();
  for (int i = 0; i < 32; i += 8)
    o[(size_t)(e0 + ty + i) * LL + (l0 + tx)] = (bf16_t)tile[tx][ty + i];
}

// ---------------------------------------------------------------------------
// Tensor Data Mover: 2D tile (rows x 32 bf16 elems) global -> LDS
// D# built per CDNA5 ISA bitfields.
// Must be executed by a single wave per workgroup; tracked by TENSORcnt.
// ---------------------------------------------------------------------------
__device__ inline void tdm_load_tile2d(const void* gptr, unsigned ldsByte,
                                       int rowLenElems /*tensor_dim0 & stride*/,
                                       int rows /*tile_dim1*/) {
  u64 ga = (u64)(size_t)gptr;
  // group0: [1:0]=count=1 | [63:32]=lds_addr | [120:64]=global_addr | [127:126]=type=2
  u32x4 g0 = { 1u,
               ldsByte,
               (u32)(ga & 0xFFFFFFFFu),
               (u32)((ga >> 32) & 0x01FFFFFFu) | 0x80000000u };
  // group1: data_size=1(2B) | tensor_dim0=K | tensor_dim1=rows | tile_dim0=32 |
  //         tile_dim1=rows | tile_dim2=0 | tensor_dim0_stride=K | tensor_dim1_stride=0
  i32x8 g1 = {
    (int)0x00010000u,                                        // [17:16]=data_size=1
    (int)(((u32)rowLenElems & 0xFFFFu) << 16),               // tensor_dim0[15:0] @ bits 63:48
    (int)((((u32)rowLenElems >> 16) & 0xFFFFu) |             // tensor_dim0[31:16]
          (((u32)rows & 0xFFFFu) << 16)),                    // tensor_dim1[15:0]
    (int)(32u << 16),                                        // tensor_dim1[31:16]=0 | tile_dim0=32
    (int)((u32)rows & 0xFFFFu),                              // tile_dim1=rows | tile_dim2=0
    (int)(u32)rowLenElems,                                   // tensor_dim0_stride[31:0]
    0,                                                       // stride hi | tensor_dim1_stride lo
    0 };
  i32x4 z4 = {0, 0, 0, 0};
#if defined(__clang_major__) && (__clang_major__ >= 23)
  i32x8 z8 = {0, 0, 0, 0, 0, 0, 0, 0};
  __builtin_amdgcn_tensor_load_to_lds(g0, g1, z4, z4, z8, 0);
#else
  __builtin_amdgcn_tensor_load_to_lds(g0, g1, z4, z4, 0);
#endif
}

// ---------------------------------------------------------------------------
// bf16 WMMA GEMM with TDM double-buffered LDS staging:
//   C[M,N] = A[M,K] * B[N,K]^T (+bias +add0 +add1)
// block tile 128(M) x 64(N) x 32(K); 8 waves -> each a 32x32 tile (2x2 WMMAs)
// wave 0 drives the Tensor Data Mover; compute reads fragments via ds_load.
// ---------------------------------------------------------------------------
// A fragment per ISA "16-bit A-Matrix 16x32" table (ld = 32 inside LDS tile)
__device__ inline bf16x16 lds_fragA(const bf16_t* tile, int row0, int lane) {
  int lr = lane & 15, hi = lane >> 4;
  const bf16_t* p = tile + (size_t)(row0 + lr) * 32 + hi * 8;
  union { bf16x16 v; uint4 u[2]; } f;
  f.u[0] = *reinterpret_cast<const uint4*>(p);
  f.u[1] = *reinterpret_cast<const uint4*>(p + 16);
  return f.v;
}
// B fragment: same K order as A table (K=8h..8h+7, 16+8h..16+8h+7) (B stored transposed as [N, K])
__device__ inline bf16x16 lds_fragB(const bf16_t* tile, int n0, int lane) {
  int ln = lane & 15, hi = lane >> 4;
  const bf16_t* p = tile + (size_t)(n0 + ln) * 32 + hi * 8;
  union { bf16x16 v; uint4 u[2]; } f;
  f.u[0] = *reinterpret_cast<const uint4*>(p);
  f.u[1] = *reinterpret_cast<const uint4*>(p + 16);
  return f.v;
}

__global__ __launch_bounds__(256) void gemm_bf16_wmma_kernel(
    const bf16_t* __restrict__ A, long long aBatch,
    const bf16_t* __restrict__ Bm, long long bBatch,
    float* __restrict__ C, long long cBatch,
    const float* __restrict__ bias,
    const float* __restrict__ add0,
    const float* __restrict__ add1,
    int M, int N, int K) {
  __shared__ bf16_t tA[2][128 * 32];  // 2 x 8 KB
  __shared__ bf16_t tB[2][64 * 32];   // 2 x 4 KB
  int z = blockIdx.z;
  const bf16_t* Ab = A + (size_t)z * aBatch;
  const bf16_t* Bb = Bm + (size_t)z * bBatch;
  size_t cOff = (size_t)z * cBatch;
  int mBase = blockIdx.y * 128;
  int nBase = blockIdx.x * 64;
  int lane = threadIdx.x & 31, wid = threadIdx.x >> 5;
  int wmLoc = (wid & 3) * 32;   // local A rows
  int wnLoc = (wid >> 2) * 32;  // local B rows
  bool issuer = (threadIdx.x < 32);  // wave 0 drives the TDM

  int nk = K >> 5;
  if (issuer) {
    tdm_load_tile2d(Ab + (size_t)mBase * K, (u32)(size_t)&tA[0][0], K, 128);
    tdm_load_tile2d(Bb + (size_t)nBase * K, (u32)(size_t)&tB[0][0], K, 64);
  }

  f32x8 acc00 = {}, acc01 = {}, acc10 = {}, acc11 = {};
  for (int kt = 0; kt < nk; ++kt) {
    if (issuer) __builtin_amdgcn_s_wait_tensorcnt(0);
    __syncthreads();  // tile kt visible to all waves; previous tile fully consumed
    int cur = kt & 1;
    if (issuer && (kt + 1 < nk)) {  // DMA next tile while computing this one
      int k0 = (kt + 1) << 5;
      tdm_load_tile2d(Ab + (size_t)mBase * K + k0, (u32)(size_t)&tA[cur ^ 1][0], K, 128);
      tdm_load_tile2d(Bb + (size_t)nBase * K + k0, (u32)(size_t)&tB[cur ^ 1][0], K, 64);
    }
    bf16x16 a0 = lds_fragA(tA[cur], wmLoc, lane);
    bf16x16 a1 = lds_fragA(tA[cur], wmLoc + 16, lane);
    bf16x16 b0 = lds_fragB(tB[cur], wnLoc, lane);
    bf16x16 b1 = lds_fragB(tB[cur], wnLoc + 16, lane);
    acc00 = __builtin_amdgcn_wmma_f32_16x16x32_bf16(false, a0, false, b0, (short)0, acc00, false, false);
    acc01 = __builtin_amdgcn_wmma_f32_16x16x32_bf16(false, a0, false, b1, (short)0, acc01, false, false);
    acc10 = __builtin_amdgcn_wmma_f32_16x16x32_bf16(false, a1, false, b0, (short)0, acc10, false, false);
    acc11 = __builtin_amdgcn_wmma_f32_16x16x32_bf16(false, a1, false, b1, (short)0, acc11, false, false);
    __syncthreads();  // all waves done with buf[cur] before TDM overwrites it next+1
  }

  int hi = lane >> 4, lr = lane & 15;
  int wm = mBase + wmLoc, wn = nBase + wnLoc;
  auto store_tile = [&](const f32x8& acc, int m0, int n0) {
    int n = n0 + lr;
    float bv = bias ? bias[n] : 0.0f;
    for (int r = 0; r < 8; ++r) {
      int m = m0 + r + 8 * hi;  // C/D layout: VGPR r -> M=r (lo lanes) / r+8 (hi)
      size_t idx = (size_t)m * N + n;
      float v = acc[r] + bv;
      if (add0) v += add0[idx];
      if (add1) v += add1[idx];
      C[cOff + idx] = v;
    }
  };
  store_tile(acc00, wm, wn);
  store_tile(acc01, wm, wn + 16);
  store_tile(acc10, wm + 16, wn);
  store_tile(acc11, wm + 16, wn + 16);
}

// ---------------------------------------------------------------------------
// LayerNorm over E, one block per (b, l) row
// ---------------------------------------------------------------------------
__global__ __launch_bounds__(256) void layernorm_kernel(const float* __restrict__ x,
                                                        const float* __restrict__ gamma,
                                                        const float* __restrict__ beta,
                                                        float* __restrict__ out) {
  __shared__ float red[256];
  const float* xr = x + (size_t)blockIdx.x * EE;
  float* orow = out + (size_t)blockIdx.x * EE;
  float s = 0.0f;
  for (int e = threadIdx.x; e < EE; e += 256) s += xr[e];
  red[threadIdx.x] = s; __syncthreads();
  for (int st = 128; st > 0; st >>= 1) {
    if (threadIdx.x < st) red[threadIdx.x] += red[threadIdx.x + st];
    __syncthreads();
  }
  float mu = red[0] * (1.0f / (float)EE);
  __syncthreads();
  float vs = 0.0f;
  for (int e = threadIdx.x; e < EE; e += 256) { float dd = xr[e] - mu; vs += dd * dd; }
  red[threadIdx.x] = vs; __syncthreads();
  for (int st = 128; st > 0; st >>= 1) {
    if (threadIdx.x < st) red[threadIdx.x] += red[threadIdx.x + st];
    __syncthreads();
  }
  float inv = rsqrtf(red[0] * (1.0f / (float)EE) + 1e-12f);
  for (int e = threadIdx.x; e < EE; e += 256)
    orow[e] = (xr[e] - mu) * inv * gamma[e] + beta[e];
}

// ---------------------------------------------------------------------------
// host launcher
// ---------------------------------------------------------------------------
extern "C" void kernel_launch(void* const* d_in, const int* in_sizes, int n_in,
                              void* d_out, int out_size, void* d_ws, size_t ws_size,
                              hipStream_t stream) {
  (void)in_sizes; (void)n_in; (void)out_size; (void)ws_size;
  const float* emb    = (const float*)d_in[0];
  const float* wPre   = (const float*)d_in[1];
  const float* bPre   = (const float*)d_in[2];
  const float* wSp    = (const float*)d_in[3];
  const float* bSp    = (const float*)d_in[4];
  const float* psfW   = (const float*)d_in[5];
  const float* wPost  = (const float*)d_in[6];
  const float* bPost  = (const float*)d_in[7];
  const float* sparse = (const float*)d_in[8];
  const float* gamma  = (const float*)d_in[9];
  const float* beta   = (const float*)d_in[10];
  float* out = (float*)d_out;

  const long long BL = (long long)BB * LL;       // 8192
  const long long WE = (long long)EE * EE;       // 1M

  char* ws = (char*)d_ws;
  size_t off = 0;
  auto wsalloc = [&](size_t bytes) -> char* {
    char* p = ws + off;
    off = (off + bytes + 255) & ~(size_t)255;
    return p;
  };

  bf16_t* embBF    = (bf16_t*)wsalloc(BL * EE * 2);
  bf16_t* wPreBF   = (bf16_t*)wsalloc(WE * 2);
  bf16_t* wSpBF    = (bf16_t*)wsalloc(WE * 2);
  bf16_t* wPostBF  = (bf16_t*)wsalloc(WE * 2);
  bf16_t* sparseBF = (bf16_t*)wsalloc((size_t)LL * LL * 2);
  float*  pre      = (float*)wsalloc(BL * EE * 4);
  bf16_t* preBF    = (bf16_t*)wsalloc(BL * EE * 2);
  float*  preSp    = (float*)wsalloc(BL * EE * 4);
  bf16_t* preSpT   = (bf16_t*)wsalloc(BL * EE * 2);          // [b][E][L]
  float*  sparseEmb= (float*)wsalloc(BL * EE * 4);
  float*  psfs     = (float*)wsalloc((size_t)HH * NTAP * DD * 4);
  c2*     psfRows  = (c2*)wsalloc((size_t)HH * NTAP * DD * 8);
  c2*     psfF     = (c2*)wsalloc((size_t)HH * NSEQ * DD * 8);
  c2*     embRows  = (c2*)wsalloc((size_t)HH * LL * DD * 8);      // per-batch reuse
  c2*     embG     = (c2*)wsalloc((size_t)HH * 2 * LL * DD * 8);  // per-batch reuse
  float*  conv     = (float*)wsalloc(BL * EE * 4);
  bf16_t* convBF   = (bf16_t*)wsalloc(BL * EE * 2);
  float*  xres     = (float*)wsalloc(BL * EE * 4);

  // 1) bf16 conversions of GEMM operands
  cvt_f32_bf16_kernel<<<1024, 256, 0, stream>>>(emb, embBF, BL * EE);
  cvt_f32_bf16_kernel<<<512, 256, 0, stream>>>(wPre, wPreBF, WE);
  cvt_f32_bf16_kernel<<<512, 256, 0, stream>>>(wSp, wSpBF, WE);
  cvt_f32_bf16_kernel<<<512, 256, 0, stream>>>(wPost, wPostBF, WE);
  cvt_tril_bf16_kernel<<<1024, 256, 0, stream>>>(sparse, sparseBF);

  // 2) PSF: masked softmax over taps, then 2D forward FFT (D axis, then seq)
  psf_softmax_kernel<<<HH * DD, 256, 0, stream>>>(psfW, psfs);
  fft64_rows_kernel<<<(HH * NTAP + 63) / 64, 64, 0, stream>>>(
      psfs, (long long)NTAP * DD, (long long)DD, psfRows, NTAP);
  psf_seqfft_kernel<<<HH * DD, 256, 0, stream>>>(psfRows, psfF);

  // 3) pre = emb @ Wpre^T + b
  gemm_bf16_wmma_kernel<<<dim3(EE / 64, BL / 128, 1), 256, 0, stream>>>(
      embBF, 0, wPreBF, 0, pre, 0, bPre, nullptr, nullptr, (int)BL, EE, EE);
  cvt_f32_bf16_kernel<<<1024, 256, 0, stream>>>(pre, preBF, BL * EE);

  // 4) pre_sparse = pre @ Wsp^T + b ; transpose to [E, L] bf16 per batch
  gemm_bf16_wmma_kernel<<<dim3(EE / 64, BL / 128, 1), 256, 0, stream>>>(
      preBF, 0, wSpBF, 0, preSp, 0, bSp, nullptr, nullptr, (int)BL, EE, EE);
  transpose_bf16_kernel<<<dim3(LL / 32, EE / 32, BB), 256, 0, stream>>>(preSp, preSpT);

  // 5) sparse_emb[b] = tril(sparse) @ pre_sparse[b]   (batched, A shared)
  gemm_bf16_wmma_kernel<<<dim3(EE / 64, LL / 128, BB), 256, 0, stream>>>(
      sparseBF, 0, preSpT, (long long)EE * LL, sparseEmb, (long long)LL * EE,
      nullptr, nullptr, nullptr, LL, EE, LL);

  // 6) FFT convolution branch, per batch (reuses embRows/embG scratch)
  for (int b = 0; b < BB; ++b) {
    const float* preb = pre + (size_t)b * LL * EE;
    fft64_rows_kernel<<<(HH * LL + 63) / 64, 64, 0, stream>>>(
        preb, (long long)DD, (long long)EE, embRows, LL);
    emb_seqfft_kernel<<<HH * DD, 256, 0, stream>>>(embRows, psfF, embG);
    idfft_fold_gelu_kernel<<<dim3(LL / 64, HH), 64, 0, stream>>>(
        embG, conv + (size_t)b * LL * EE);
  }
  cvt_f32_bf16_kernel<<<1024, 256, 0, stream>>>(conv, convBF, BL * EE);

  // 7) x = conv @ Wpost^T + b + sparse_emb + embeddings (fused epilogue)
  gemm_bf16_wmma_kernel<<<dim3(EE / 64, BL / 128, 1), 256, 0, stream>>>(
      convBF, 0, wPostBF, 0, xres, 0, bPost, sparseEmb, emb, (int)BL, EE, EE);

  // 8) LayerNorm
  layernorm_kernel<<<(unsigned)BL, 256, 0, stream>>>(xres, gamma, beta, out);
}
